// DeformableAttention1D_47158740910186
// MI455X (gfx1250) — compile-verified, matrix operand order checked by reading, not run on hardware
//
#include <hip/hip_runtime.h>
#include <hip/hip_bf16.h>

// ---------------------------------------------------------------------------
// Problem constants (match the JAX reference)
// ---------------------------------------------------------------------------
constexpr int BB   = 256;   // batch
constexpr int DIMC = 384;   // model dim
constexpr int NN   = 128;   // sequence length
constexpr int HH   = 4;     // heads
constexpr int GG   = 2;     // offset groups
constexpr int NKK  = 4;     // sampled keys per query
constexpr int DHH  = 96;    // head dim
constexpr int DGG  = 192;   // dim per offset group
constexpr int KSS  = 7;     // depthwise kernel size
constexpr int DFFC = 2048;  // FFN hidden
constexpr float EPSC = 1e-5f;

typedef __attribute__((ext_vector_type(16))) __bf16 bf16x16;
typedef __attribute__((ext_vector_type(8)))  float  f32x8;

// gfx1250 async global->LDS copy (ASYNCcnt path), guarded so the file still
// compiles if this toolchain does not expose the builtins.
#if defined(__HIP_DEVICE_COMPILE__) && \
    __has_builtin(__builtin_amdgcn_global_load_async_to_lds_b128) && \
    __has_builtin(__builtin_amdgcn_s_wait_asynccnt)
#define DEFORM_ASYNC_LDS 1
// Builtin signature (from the round-3 diagnostic): param0 is a non-const
// pointer to v4i32 (vector_size(16) int) in the global address space,
// param1 the LDS-side v4i32 pointer, then imm offset and imm cpol.
typedef int deform_v4i __attribute__((vector_size(16)));
typedef __attribute__((address_space(1))) deform_v4i* deform_gv4p;
typedef __attribute__((address_space(3))) deform_v4i* deform_lv4p;
#else
#define DEFORM_ASYNC_LDS 0
#endif

// ---------------------------------------------------------------------------
// LayerNorm of positional embedding rows: pe[n][d] = LN(emb[n])[d]*g + b
// ---------------------------------------------------------------------------
__global__ __launch_bounds__(128) void deform_pe_ln(
    const float* __restrict__ emb, const float* __restrict__ g,
    const float* __restrict__ b, float* __restrict__ pe)
{
  __shared__ float red[2];
  const int n = blockIdx.x;
  const float* row = emb + (size_t)n * DIMC;
  float s = 0.f, s2 = 0.f;
  for (int d = threadIdx.x; d < DIMC; d += 128) { float x = row[d]; s += x; s2 += x * x; }
  if (threadIdx.x < 2) red[threadIdx.x] = 0.f;
  __syncthreads();
  atomicAdd(&red[0], s); atomicAdd(&red[1], s2);
  __syncthreads();
  const float mean = red[0] / DIMC;
  const float inv  = rsqrtf(red[1] / DIMC - mean * mean + EPSC);
  for (int d = threadIdx.x; d < DIMC; d += 128)
    pe[(size_t)n * DIMC + d] = (row[d] - mean) * inv * g[d] + b[d];
}

// ---------------------------------------------------------------------------
// raw(B,DIM,N) + pe(N,DIM) -> pos f32 (channel-major) and optional bf16
// token-major activations tok[(b*N+n)*DIM + d]
// ---------------------------------------------------------------------------
__global__ void deform_add_pe(const float* __restrict__ raw, const float* __restrict__ pe,
                              float* __restrict__ pos, __bf16* __restrict__ tok)
{
  const size_t total = (size_t)BB * DIMC * NN;
  size_t i = (size_t)blockIdx.x * 256 + threadIdx.x;
  if (i >= total) return;
  const int n = (int)(i % NN);
  const size_t t = i / NN;
  const int d = (int)(t % DIMC);
  const size_t b = t / DIMC;
  const float v = raw[i] + pe[(size_t)n * DIMC + d];
  pos[i] = v;
  if (tok) tok[(b * NN + n) * (size_t)DIMC + d] = (__bf16)v;
}

// ---------------------------------------------------------------------------
// Weight prep: f32 W[K][Nd] -> bf16 Wt[Nd][K] (transposed once; weights are
// loop-invariant and L2-resident, so the GEMM stages B tiles with plain
// contiguous b128 copies).
// ---------------------------------------------------------------------------
__global__ void deform_f2bf_t(const float* __restrict__ x, __bf16* __restrict__ y,
                              int K, int Nd)
{
  int i = blockIdx.x * 256 + threadIdx.x;
  if (i >= K * Nd) return;
  const int n = i % Nd;
  const int k = i / Nd;
  y[(size_t)n * K + k] = (__bf16)x[i];
}

// ---------------------------------------------------------------------------
// WMMA bf16 GEMM:  Y[M,Nd] = epilogue(X[M,K] @ W[K,Nd] + bias)
// 128x128 output tile per 256-thread workgroup (8 waves). Each wave computes
// a 32x64 strip: 2 A-fragments x 4 B-fragments -> 8 v_wmma_f32_16x16x32_bf16
// per K-step of 32. LDS tiles (rows x 32) bf16, rows padded to 40 elements
// (80 B) for 16B-aligned, bank-spread half-wave fragment reads. Tiles are
// staged with gfx1250 async global->LDS copies when available.
// ---------------------------------------------------------------------------
template<bool RELU, bool OUT_BF16, bool ADD_RES>
__global__ __launch_bounds__(256) void deform_gemm_bf16(
    const __bf16* __restrict__ X, const __bf16* __restrict__ Wt,  // Wt[Nd][K]
    const float* __restrict__ bias, const float* __restrict__ res,
    __bf16* __restrict__ Ybf, float* __restrict__ Yf,
    int M, int K, int Nd)
{
  __shared__ __bf16 As[128][40];   // A tile rows (m, k)
  __shared__ __bf16 Bs[128][40];   // B tile rows (n, k)  (from transposed W)

  const int tid  = threadIdx.x;
  const int lane = tid & 31;
  const int wave = tid >> 5;       // 0..7
  const int wm   = wave >> 1;      // 0..3 : 32-row strip
  const int wn   = wave & 1;       // 0..1 : 64-col strip
  const int m0 = blockIdx.x * 128;
  const int n0 = blockIdx.y * 128;

  f32x8 acc[2][4] = {};

  const int rr = tid >> 1;         // 0..127 staging row
  const int cc = (tid & 1) * 16;   // 0 / 16 staging col

  for (int k0 = 0; k0 < K; k0 += 32) {
    const __bf16* gx = X  + (size_t)(m0 + rr) * K + (k0 + cc);
    const __bf16* gw = Wt + (size_t)(n0 + rr) * K + (k0 + cc);
#if DEFORM_ASYNC_LDS
    {
      // ASYNCcnt-tracked DMA into LDS; inst offset applies to both addresses.
      deform_gv4p gxp = (deform_gv4p)(unsigned long long)(const void*)gx;
      deform_gv4p gwp = (deform_gv4p)(unsigned long long)(const void*)gw;
      deform_lv4p lap = (deform_lv4p)(unsigned)(unsigned long long)(void*)&As[rr][cc];
      deform_lv4p lbp = (deform_lv4p)(unsigned)(unsigned long long)(void*)&Bs[rr][cc];
      __builtin_amdgcn_global_load_async_to_lds_b128(gxp, lap, 0, 0);
      __builtin_amdgcn_global_load_async_to_lds_b128(gxp, lap, 16, 0);
      __builtin_amdgcn_global_load_async_to_lds_b128(gwp, lbp, 0, 0);
      __builtin_amdgcn_global_load_async_to_lds_b128(gwp, lbp, 16, 0);
    }
#else
    {
      // Fallback: load all four 16B chunks into registers first so the
      // global loads form one clause with a single wait, then store to LDS.
      const uint4* vx = (const uint4*)gx;
      const uint4* vw = (const uint4*)gw;
      uint4 xa = vx[0], xb = vx[1];
      uint4 wa = vw[0], wb = vw[1];
      uint4* la = (uint4*)&As[rr][cc];
      uint4* lb = (uint4*)&Bs[rr][cc];
      la[0] = xa; la[1] = xb;
      lb[0] = wa; lb[1] = wb;
    }
#endif
    if (k0 + 32 < K) {  // gfx1250 global_prefetch of next tiles
      __builtin_prefetch(gx + 32, 0, 1);
      __builtin_prefetch(gw + 32, 0, 1);
    }
#if DEFORM_ASYNC_LDS
    __builtin_amdgcn_s_wait_asynccnt(0);
#endif
    __syncthreads();

    // Fragment packing per CDNA5 ISA 7.12.2: lane L holds row (L&15);
    // lanes 0-15 carry K={0..7,16..23}, lanes 16-31 K={8..15,24..31}.
    const int half8 = ((lane >> 4) & 1) * 8;
    const int rowin = lane & 15;
    bf16x16 a[2];
    #pragma unroll
    for (int ms = 0; ms < 2; ++ms) {
      const __bf16* pa = &As[wm * 32 + ms * 16 + rowin][half8];
      __builtin_memcpy(&a[ms], pa, 16);
      __builtin_memcpy(((char*)&a[ms]) + 16, pa + 16, 16);
    }
    #pragma unroll
    for (int ns = 0; ns < 4; ++ns) {
      bf16x16 bfr;
      const __bf16* pb = &Bs[wn * 64 + ns * 16 + rowin][half8];
      __builtin_memcpy(&bfr, pb, 16);
      __builtin_memcpy(((char*)&bfr) + 16, pb + 16, 16);
      #pragma unroll
      for (int ms = 0; ms < 2; ++ms)
        acc[ms][ns] = __builtin_amdgcn_wmma_f32_16x16x32_bf16(
            false, a[ms], false, bfr, (short)0, acc[ms][ns], false, false);
    }
    __syncthreads();
  }

  // Epilogue: C/D layout — lane L, VGPR j => m = j + 8*(L>=16), n = L&15
  const int half = (lane >> 4) & 1;
  const int ncol = lane & 15;
  #pragma unroll
  for (int ms = 0; ms < 2; ++ms) {
    #pragma unroll
    for (int ns = 0; ns < 4; ++ns) {
      #pragma unroll
      for (int j = 0; j < 8; ++j) {
        const int m = wm * 32 + ms * 16 + half * 8 + j;
        const int n = wn * 64 + ns * 16 + ncol;
        const size_t gn  = (size_t)(n0 + n);
        const size_t idx = (size_t)(m0 + m) * Nd + gn;
        float y = acc[ms][ns][j] + bias[gn];
        if (RELU)    y = fmaxf(y, 0.f);
        if (ADD_RES) y += res[idx];
        if (OUT_BF16) Ybf[idx] = (__bf16)y;
        else          Yf[idx]  = y;
      }
    }
  }
}

// ---------------------------------------------------------------------------
// Offset head: depthwise conv7 + GELU(exact) + pointwise(NK) + tanh*4
// One workgroup per (b,g); thread = position n; channels chunked through LDS.
// Qt is token-major f32 [(b*N+n)*DIM + d].  Emits sampling coords iy.
// ---------------------------------------------------------------------------
__global__ __launch_bounds__(128) void deform_offsets(
    const float* __restrict__ Qt, const float* __restrict__ dw_w,
    const float* __restrict__ dw_b, const float* __restrict__ pw_w,
    float* __restrict__ iy_out)
{
  constexpr int CHUNK = 48;
  __shared__ float tile[CHUNK][NN + 8];
  const int bg = blockIdx.x;
  const int b = bg / GG, g = bg % GG;
  const int n = threadIdx.x;
  float acc[NKK] = {0.f, 0.f, 0.f, 0.f};

  for (int c0 = 0; c0 < DGG; c0 += CHUNK) {
    const float* src = Qt + ((size_t)(b * NN + n)) * DIMC + g * DGG + c0;
    for (int i = 0; i < CHUNK; ++i) tile[i][n] = src[i];
    __syncthreads();
    for (int i = 0; i < CHUNK; ++i) {
      const int c = c0 + i;
      float s = dw_b[c];
      #pragma unroll
      for (int t = 0; t < KSS; ++t) {
        const int nn = n + t - (KSS - 1) / 2;
        const float xv = (nn >= 0 && nn < NN) ? tile[i][nn] : 0.f;
        s += dw_w[c * KSS + t] * xv;
      }
      const float gel = 0.5f * s * (1.f + erff(s * 0.70710678118654752f));
      #pragma unroll
      for (int k = 0; k < NKK; ++k) acc[k] += pw_w[k * DGG + c] * gel;
    }
    __syncthreads();
  }
  #pragma unroll
  for (int k = 0; k < NKK; ++k) {
    const float off   = tanhf(acc[k]) * 4.0f;
    const float vgrid = (float)n + off;
    float vg = 2.f * vgrid / (float)(NN - 1) - 1.f;
    vg = fminf(1.f, fmaxf(-1.f, vg));
    iy_out[((size_t)bg * NKK + k) * NN + n] = ((vg + 1.f) * NN - 1.f) * 0.5f;
  }
}

// ---------------------------------------------------------------------------
// Bilinear sampling (zero pad) -> bf16 kv features, token-major
// Xkv[((b*NK+k)*N+n)*DIM + d] with d = g*DG + c
// ---------------------------------------------------------------------------
__global__ void deform_sample(const float* __restrict__ kv_pos,
                              const float* __restrict__ iy_buf,
                              __bf16* __restrict__ Xkv)
{
  const size_t total = (size_t)BB * NKK * NN * DIMC;
  size_t i = (size_t)blockIdx.x * 256 + threadIdx.x;
  if (i >= total) return;
  const int d = (int)(i % DIMC);
  size_t t = i / DIMC;
  const int n = (int)(t % NN);  t /= NN;
  const int k = (int)(t % NKK);
  const int b = (int)(t / NKK);
  const int g = d / DGG;
  const int bg = b * GG + g;

  const float iy  = iy_buf[((size_t)bg * NKK + k) * NN + n];
  const float y0f = floorf(iy);
  const float w   = iy - y0f;
  const int y0 = (int)y0f;
  const int y1 = y0 + 1;
  const float* chan = kv_pos + ((size_t)b * DIMC + d) * NN;
  const float v0 = (y0 >= 0 && y0 < NN) ? chan[y0] : 0.f;
  const float v1 = (y1 >= 0 && y1 < NN) ? chan[y1] : 0.f;
  Xkv[i] = (__bf16)(v0 * (1.f - w) + v1 * w);
}

// ---------------------------------------------------------------------------
// Per-(b,n) attention over NK=4 sampled keys + residual + LayerNorm1.
// Emits res_f (f32, for final residual) and res_bf (bf16, FFN input).
// ---------------------------------------------------------------------------
__global__ __launch_bounds__(128) void deform_attn(
    const float* __restrict__ Qt, const __bf16* __restrict__ Kb,
    const __bf16* __restrict__ Vb, const float* __restrict__ q_pos,
    const float* __restrict__ g1, const float* __restrict__ b1,
    const float* __restrict__ attn_scale,
    float* __restrict__ res_f, __bf16* __restrict__ res_bf)
{
  __shared__ float sim[HH * NKK];
  __shared__ float attnw[HH * NKK];
  __shared__ float red[2];
  const int bn = blockIdx.x;
  const int b = bn / NN, n = bn % NN;
  const int tid = threadIdx.x;
  if (tid < HH * NKK) sim[tid] = 0.f;
  if (tid < 2) red[tid] = 0.f;
  __syncthreads();

  const float qscale = rsqrtf((float)(DHH * NN));  // DH^-0.5 * N^-0.5
  const size_t qbase = (size_t)bn * DIMC;
  float vreg[3][NKK];
  float resv[3];

  #pragma unroll
  for (int r = 0; r < 3; ++r) {
    const int d = tid + r * 128;
    const int h = d / DHH;
    const float qv = Qt[qbase + d] * qscale;
    #pragma unroll
    for (int j = 0; j < NKK; ++j) {
      const size_t kidx = ((size_t)(b * NKK + j) * NN + n) * DIMC + d;
      atomicAdd(&sim[h * NKK + j], qv * (float)Kb[kidx]);
      vreg[r][j] = (float)Vb[kidx];
    }
  }
  __syncthreads();

  if (tid < HH) {
    const float sc = attn_scale[0];
    float m = sim[tid * NKK];
    #pragma unroll
    for (int j = 1; j < NKK; ++j) m = fmaxf(m, sim[tid * NKK + j]);
    float e[NKK]; float ssum = 0.f;
    #pragma unroll
    for (int j = 0; j < NKK; ++j) { e[j] = expf((sim[tid * NKK + j] - m) * sc); ssum += e[j]; }
    #pragma unroll
    for (int j = 0; j < NKK; ++j) attnw[tid * NKK + j] = e[j] / ssum;
  }
  __syncthreads();

  float s = 0.f, s2 = 0.f;
  #pragma unroll
  for (int r = 0; r < 3; ++r) {
    const int d = tid + r * 128;
    const int h = d / DHH;
    float o = 0.f;
    #pragma unroll
    for (int j = 0; j < NKK; ++j) o += attnw[h * NKK + j] * vreg[r][j];
    const float rv = q_pos[((size_t)b * DIMC + d) * NN + n] + o;
    resv[r] = rv; s += rv; s2 += rv * rv;
  }
  atomicAdd(&red[0], s); atomicAdd(&red[1], s2);
  __syncthreads();
  const float mean = red[0] / DIMC;
  const float inv  = rsqrtf(red[1] / DIMC - mean * mean + EPSC);
  #pragma unroll
  for (int r = 0; r < 3; ++r) {
    const int d = tid + r * 128;
    const float y = (resv[r] - mean) * inv * g1[d] + b1[d];
    res_f[qbase + d]  = y;
    res_bf[qbase + d] = (__bf16)y;
  }
}

// ---------------------------------------------------------------------------
// Final LayerNorm2 + transpose back to (B, DIM, N)
// ---------------------------------------------------------------------------
__global__ __launch_bounds__(128) void deform_final_ln(
    const float* __restrict__ res2, const float* __restrict__ g2,
    const float* __restrict__ b2, float* __restrict__ out)
{
  __shared__ float red[2];
  const int bn = blockIdx.x;
  const int b = bn / NN, n = bn % NN;
  const int tid = threadIdx.x;
  if (tid < 2) red[tid] = 0.f;
  __syncthreads();
  float x[3]; float s = 0.f, s2 = 0.f;
  #pragma unroll
  for (int r = 0; r < 3; ++r) {
    x[r] = res2[(size_t)bn * DIMC + tid + r * 128];
    s += x[r]; s2 += x[r] * x[r];
  }
  atomicAdd(&red[0], s); atomicAdd(&red[1], s2);
  __syncthreads();
  const float mean = red[0] / DIMC;
  const float inv  = rsqrtf(red[1] / DIMC - mean * mean + EPSC);
  #pragma unroll
  for (int r = 0; r < 3; ++r) {
    const int d = tid + r * 128;
    out[((size_t)b * DIMC + d) * NN + n] = (x[r] - mean) * inv * g2[d] + b2[d];
  }
}

// ---------------------------------------------------------------------------
// Host orchestration
// ---------------------------------------------------------------------------
extern "C" void kernel_launch(void* const* d_in, const int* in_sizes, int n_in,
                              void* d_out, int out_size, void* d_ws, size_t ws_size,
                              hipStream_t stream)
{
  (void)in_sizes; (void)n_in; (void)out_size; (void)ws_size;
  const float* q_raw   = (const float*)d_in[0];
  const float* kv_raw  = (const float*)d_in[1];
  const float* emb_q   = (const float*)d_in[2];
  const float* ln_pq_g = (const float*)d_in[3];
  const float* ln_pq_b = (const float*)d_in[4];
  const float* emb_kv  = (const float*)d_in[5];
  const float* ln_pkv_g= (const float*)d_in[6];
  const float* ln_pkv_b= (const float*)d_in[7];
  const float* Wq1 = (const float*)d_in[8];  const float* bq1 = (const float*)d_in[9];
  const float* Wq2 = (const float*)d_in[10]; const float* bq2 = (const float*)d_in[11];
  const float* Wk1 = (const float*)d_in[12]; const float* bk1 = (const float*)d_in[13];
  const float* Wk2 = (const float*)d_in[14]; const float* bk2 = (const float*)d_in[15];
  const float* Wv1 = (const float*)d_in[16]; const float* bv1 = (const float*)d_in[17];
  const float* Wv2 = (const float*)d_in[18]; const float* bv2 = (const float*)d_in[19];
  const float* dw_w = (const float*)d_in[20]; const float* dw_b = (const float*)d_in[21];
  const float* pw_w = (const float*)d_in[22];
  const float* ln1_g = (const float*)d_in[23]; const float* ln1_b = (const float*)d_in[24];
  const float* ln2_g = (const float*)d_in[25]; const float* ln2_b = (const float*)d_in[26];
  const float* Wf1 = (const float*)d_in[27]; const float* bf1 = (const float*)d_in[28];
  const float* Wf2 = (const float*)d_in[29]; const float* bf2 = (const float*)d_in[30];
  const float* attn_scale = (const float*)d_in[31];
  float* outp = (float*)d_out;

  char* ws = (char*)d_ws;
  size_t o = 0;
  auto take = [&](size_t nbytes) { size_t r = o; o += (nbytes + 255) & ~(size_t)255; return r; };

  const size_t TOK_Q  = (size_t)BB * NN;        // 32768 query tokens
  const size_t TOK_KV = (size_t)BB * NKK * NN;  // 131072 kv tokens

  // weights (bf16 transposed copies) + pe
  const size_t o_peq  = take((size_t)NN * DIMC * 4);
  const size_t o_pekv = take((size_t)NN * DIMC * 4);
  const size_t o_wq1 = take((size_t)DIMC * DIMC * 2);
  const size_t o_wq2 = take((size_t)DIMC * DIMC * 2);
  const size_t o_wk1 = take((size_t)DIMC * DIMC * 2);
  const size_t o_wk2 = take((size_t)DIMC * DIMC * 2);
  const size_t o_wv1 = take((size_t)DIMC * DIMC * 2);
  const size_t o_wv2 = take((size_t)DIMC * DIMC * 2);
  const size_t o_wf1 = take((size_t)DIMC * DFFC * 2);
  const size_t o_wf2 = take((size_t)DFFC * DIMC * 2);
  // activations (lifetime-aliased regions)
  const size_t o_qpos  = take((size_t)BB * DIMC * NN * 4);                 // live through attn
  const size_t o_qt    = take(TOK_Q * DIMC * 4);                           // q (f32 token-major)
  const size_t o_iy    = take((size_t)BB * GG * NKK * NN * 4);
  const size_t o_kvpos = take((size_t)BB * DIMC * NN * 4);                 // dies after sample
  const size_t o_resf  = o_kvpos;                                          // reuse -> LN1 out f32
  const size_t o_xq    = take(TOK_Q * DIMC * 2);                           // dies after GEMM1
  const size_t o_resbf = o_xq;                                             // reuse -> LN1 out bf16
  const size_t o_h1    = take(TOK_Q * DIMC * 2);
  const size_t o_xkv   = take(TOK_KV * DIMC * 2);                          // dies after GEMM3/4
  const size_t o_hk    = take(TOK_KV * DIMC * 2);                          // dies after GEMM5
  const size_t o_hf    = o_xkv;                                            // reuse -> FFN hidden
  const size_t o_hv    = take(TOK_KV * DIMC * 2);                          // dies after GEMM6
  const size_t o_res2  = o_hv;                                             // reuse -> FFN out f32
  const size_t o_kb    = take(TOK_KV * DIMC * 2);
  const size_t o_vb    = take(TOK_KV * DIMC * 2);

  // 1) positional encodings + add
  deform_pe_ln<<<NN, 128, 0, stream>>>(emb_q,  ln_pq_g,  ln_pq_b,  (float*)(ws + o_peq));
  deform_pe_ln<<<NN, 128, 0, stream>>>(emb_kv, ln_pkv_g, ln_pkv_b, (float*)(ws + o_pekv));
  {
    const size_t total = (size_t)BB * DIMC * NN;
    const int blocks = (int)((total + 255) / 256);
    deform_add_pe<<<blocks, 256, 0, stream>>>(q_raw,  (float*)(ws + o_peq),
                                              (float*)(ws + o_qpos), (__bf16*)(ws + o_xq));
    deform_add_pe<<<blocks, 256, 0, stream>>>(kv_raw, (float*)(ws + o_pekv),
                                              (float*)(ws + o_kvpos), nullptr);
  }
  // 2) weight conversion + transpose (W[K][Nd] -> bf16 Wt[Nd][K])
  auto cvt = [&](const float* src, size_t dst, int K, int Nd) {
    deform_f2bf_t<<<(K * Nd + 255) / 256, 256, 0, stream>>>(src, (__bf16*)(ws + dst), K, Nd);
  };
  cvt(Wq1, o_wq1, DIMC, DIMC); cvt(Wq2, o_wq2, DIMC, DIMC);
  cvt(Wk1, o_wk1, DIMC, DIMC); cvt(Wk2, o_wk2, DIMC, DIMC);
  cvt(Wv1, o_wv1, DIMC, DIMC); cvt(Wv2, o_wv2, DIMC, DIMC);
  cvt(Wf1, o_wf1, DIMC, DFFC); cvt(Wf2, o_wf2, DFFC, DIMC);

  const dim3 blk(256);
  // 3) q-MLP
  deform_gemm_bf16<true, true, false><<<dim3((unsigned)(TOK_Q / 128), DIMC / 128), blk, 0, stream>>>(
      (const __bf16*)(ws + o_xq), (const __bf16*)(ws + o_wq1), bq1, nullptr,
      (__bf16*)(ws + o_h1), nullptr, (int)TOK_Q, DIMC, DIMC);
  deform_gemm_bf16<false, false, false><<<dim3((unsigned)(TOK_Q / 128), DIMC / 128), blk, 0, stream>>>(
      (const __bf16*)(ws + o_h1), (const __bf16*)(ws + o_wq2), bq2, nullptr,
      nullptr, (float*)(ws + o_qt), (int)TOK_Q, DIMC, DIMC);
  // 4) offsets + sampling
  deform_offsets<<<BB * GG, 128, 0, stream>>>(
      (const float*)(ws + o_qt), dw_w, dw_b, pw_w, (float*)(ws + o_iy));
  {
    const size_t total = TOK_KV * DIMC;
    deform_sample<<<(int)((total + 255) / 256), 256, 0, stream>>>(
        (const float*)(ws + o_kvpos), (const float*)(ws + o_iy), (__bf16*)(ws + o_xkv));
  }
  // 5) k/v MLPs
  deform_gemm_bf16<true, true, false><<<dim3((unsigned)(TOK_KV / 128), DIMC / 128), blk, 0, stream>>>(
      (const __bf16*)(ws + o_xkv), (const __bf16*)(ws + o_wk1), bk1, nullptr,
      (__bf16*)(ws + o_hk), nullptr, (int)TOK_KV, DIMC, DIMC);
  deform_gemm_bf16<true, true, false><<<dim3((unsigned)(TOK_KV / 128), DIMC / 128), blk, 0, stream>>>(
      (const __bf16*)(ws + o_xkv), (const __bf16*)(ws + o_wv1), bv1, nullptr,
      (__bf16*)(ws + o_hv), nullptr, (int)TOK_KV, DIMC, DIMC);
  deform_gemm_bf16<false, true, false><<<dim3((unsigned)(TOK_KV / 128), DIMC / 128), blk, 0, stream>>>(
      (const __bf16*)(ws + o_hk), (const __bf16*)(ws + o_wk2), bk2, nullptr,
      (__bf16*)(ws + o_kb), nullptr, (int)TOK_KV, DIMC, DIMC);
  deform_gemm_bf16<false, true, false><<<dim3((unsigned)(TOK_KV / 128), DIMC / 128), blk, 0, stream>>>(
      (const __bf16*)(ws + o_hv), (const __bf16*)(ws + o_wv2), bv2, nullptr,
      (__bf16*)(ws + o_vb), nullptr, (int)TOK_KV, DIMC, DIMC);
  // 6) attention + residual + LN1
  deform_attn<<<(int)TOK_Q, 128, 0, stream>>>(
      (const float*)(ws + o_qt), (const __bf16*)(ws + o_kb), (const __bf16*)(ws + o_vb),
      (const float*)(ws + o_qpos), ln1_g, ln1_b, attn_scale,
      (float*)(ws + o_resf), (__bf16*)(ws + o_resbf));
  // 7) FFN (second GEMM fuses the residual add)
  deform_gemm_bf16<true, true, false><<<dim3((unsigned)(TOK_Q / 128), DFFC / 128), blk, 0, stream>>>(
      (const __bf16*)(ws + o_resbf), (const __bf16*)(ws + o_wf1), bf1, nullptr,
      (__bf16*)(ws + o_hf), nullptr, (int)TOK_Q, DIMC, DFFC);
  deform_gemm_bf16<false, false, true><<<dim3((unsigned)(TOK_Q / 128), DIMC / 128), blk, 0, stream>>>(
      (const __bf16*)(ws + o_hf), (const __bf16*)(ws + o_wf2), bf2,
      (const float*)(ws + o_resf), nullptr, (float*)(ws + o_res2), (int)TOK_Q, DFFC, DIMC);
  // 8) final LN + transpose to (B, DIM, N)
  deform_final_ln<<<(int)TOK_Q, 128, 0, stream>>>(
      (const float*)(ws + o_res2), ln2_g, ln2_b, outp);
}
